// ModulatedDeformConv2dFunction_3504693314064
// MI455X (gfx1250) — matrix operand order repaired and checked by reading, not compiled, not run on hardware
//
#include <hip/hip_runtime.h>
#include <hip/hip_bf16.h>

typedef _Float16 v16h __attribute__((ext_vector_type(16)));
typedef float    v8f  __attribute__((ext_vector_type(8)));

#define IH 128
#define IW 128
#define HW 16384          // 128*128
#define CIN 64
#define COUT 64
#define TAPS 9
#define PTILE 64

// Dynamic LDS layout (bytes):
//   [0, 16384)      col double buffer: 2 bufs x 8 frags x 32 lanes x 16 halves
//   [16384, 90112)  weight frags: 9 taps x 4 ct x 2 kc x 32 lanes x 16 halves
#define COL_BYTES 16384
#define LDS_BYTES 90112

__global__ __launch_bounds__(256)
void dcnv2_wmma_kernel(const float* __restrict__ xin,
                       const float* __restrict__ offs,
                       const float* __restrict__ msk,
                       const float* __restrict__ wgt,
                       const float* __restrict__ bias,
                       float* __restrict__ out)
{
    extern __shared__ char smem[];
    _Float16* colH = (_Float16*)smem;                // col fragments (halves)
    _Float16* wH   = (_Float16*)(smem + COL_BYTES);  // weight fragments (halves)

    const int tid  = threadIdx.x;
    const int lane = tid & 31;
    const int wave = tid >> 5;

    const int b  = blockIdx.x >> 8;          // 256 position tiles per image
    const int p0 = (blockIdx.x & 255) * PTILE;

    // ---- stage weights (fp32 -> f16) into LDS in WMMA A-fragment layout ----
    // A 16x32 f16 layout: lane L -> g=L/16, m=L%16 ; half j -> k = derived.
    // Inverse: for k (0..31): g=k>>4, j=k&15, lane=16g+m.
    for (int idx = tid; idx < COUT * CIN * TAPS; idx += 256) {
        int tap  = idx % TAPS;
        int cin  = (idx / TAPS) % CIN;
        int cout = idx / (TAPS * CIN);
        int ct = cout >> 4, m = cout & 15;
        int kc = cin >> 5, kk = cin & 31;
        int g  = kk >> 4;
        int j  = kk & 15;
        int laneA = (g << 4) | m;
        wH[(((tap << 3) + (ct << 1) + kc) << 9) + (laneA << 4) + j] =
            (_Float16)wgt[idx];
    }

    // ---- sampling role: thread -> (pixel pl, channel group cg of 16) ----
    const int pl = tid & 63;
    const int cg = tid >> 6;
    const int p  = p0 + pl;
    const int ho = p >> 7, wo = p & 127;
    const float* xb = xin + (((size_t)b * CIN + (size_t)cg * 16) << 14);

    // B-fragment destination for this thread's 16 contiguous halves:
    // for c = cg*16+i : kc = cg>>1, k-in-frag kk = (cg&1)*16+i
    //   -> lane = 16*(kk/16) + (pl%16) = 16*(cg&1) + pl%16 (thread-constant)
    //   -> half index j = kk%16 = i (contiguous!)
    const int fragIdx = ((pl >> 4) << 1) | (cg >> 1);       // ptile*2 + kc
    const int laneF   = ((cg & 1) << 4) | (pl & 15);

    // ---- wmma role: wave -> cout tile ct, pixel tiles ptA / ptB ----
    const int ct  = wave & 3;
    const int ptA = wave >> 2;
    const int ptB = ptA + 2;
    v8f acc0 = {}, acc1 = {};

    for (int tap = 0; tap < TAPS; ++tap) {
        const int kh = tap / 3, kw = tap - kh * 3;

        // offsets/mask (layout: [B, 2*KK, Ho, Wo] / [B, KK, Ho, Wo])
        float dy = offs[(((size_t)b * 18 + 2 * tap)     << 14) + p];
        float dx = offs[(((size_t)b * 18 + 2 * tap + 1) << 14) + p];
        float mv = msk [(((size_t)b * TAPS + tap)       << 14) + p];

        float yf = (float)(ho - 1 + kh) + dy;
        float xf = (float)(wo - 1 + kw) + dx;
        float y0f = floorf(yf), x0f = floorf(xf);
        float ly = yf - y0f,    lx = xf - x0f;
        int y0 = (int)y0f, x0 = (int)x0f;
        int y1 = y0 + 1,   x1 = x0 + 1;

        // fold mask + zero-pad validity into the 4 bilinear corner weights
        float vy0 = (y0 >= 0 && y0 < IH) ? 1.f : 0.f;
        float vy1 = (y1 >= 0 && y1 < IH) ? 1.f : 0.f;
        float vx0 = (x0 >= 0 && x0 < IW) ? 1.f : 0.f;
        float vx1 = (x1 >= 0 && x1 < IW) ? 1.f : 0.f;
        float w00 = (1.f - ly) * (1.f - lx) * mv * vy0 * vx0;
        float w01 = (1.f - ly) * lx         * mv * vy0 * vx1;
        float w10 = ly * (1.f - lx)         * mv * vy1 * vx0;
        float w11 = ly * lx                 * mv * vy1 * vx1;

        int cy0 = min(max(y0, 0), IH - 1), cy1 = min(max(y1, 0), IH - 1);
        int cx0 = min(max(x0, 0), IW - 1), cx1 = min(max(x1, 0), IW - 1);
        int i00 = cy0 * IW + cx0, i01 = cy0 * IW + cx1;
        int i10 = cy1 * IW + cx0, i11 = cy1 * IW + cx1;

        // sample 16 channels, pack f32 pairs -> f16x2 (v_cvt_pk_rtz_f16_f32)
        unsigned pk[8];
#pragma unroll
        for (int ii = 0; ii < 8; ++ii) {
            const float* pc0 = xb + ((size_t)(2 * ii)     << 14);
            const float* pc1 = xb + ((size_t)(2 * ii + 1) << 14);
            float v0 = w00 * pc0[i00] + w01 * pc0[i01]
                     + w10 * pc0[i10] + w11 * pc0[i11];
            float v1 = w00 * pc1[i00] + w01 * pc1[i01]
                     + w10 * pc1[i10] + w11 * pc1[i11];
            auto h2 = __builtin_amdgcn_cvt_pkrtz(v0, v1);
            pk[ii] = __builtin_bit_cast(unsigned, h2);
        }
        // two ds_store_b128 straight into B-fragment layout (double buffered)
        char* dst = (char*)colH + ((tap & 1) << 13) + (fragIdx << 10) + (laneF << 5);
        ((uint4*)dst)[0] = make_uint4(pk[0], pk[1], pk[2], pk[3]);
        ((uint4*)dst)[1] = make_uint4(pk[4], pk[5], pk[6], pk[7]);

        __syncthreads();   // one barrier per tap (col is double buffered)

        const _Float16* buf = colH + ((tap & 1) << 12);  // halves
#pragma unroll
        for (int kc = 0; kc < 2; ++kc) {
            v16h a  = *(const v16h*)(wH +
                        ((((tap << 3) + (ct << 1) + kc) << 9) + (lane << 4)));
            v16h b0 = *(const v16h*)(buf +
                        ((((ptA << 1) + kc) << 9) + (lane << 4)));
            acc0 = __builtin_amdgcn_wmma_f32_16x16x32_f16(
                       false, a, false, b0, (short)0, acc0, false, false);
            v16h b1 = *(const v16h*)(buf +
                        ((((ptB << 1) + kc) << 9) + (lane << 4)));
            acc1 = __builtin_amdgcn_wmma_f32_16x16x32_f16(
                       false, a, false, b1, (short)0, acc1, false, false);
        }
    }

    // ---- epilogue: C layout -> global. C[m][n]: m = r + 8*(lane/16), n = lane%16
    const int n  = lane & 15;
    const int gg = lane >> 4;
    float* outB = out + ((size_t)b << 20);   // b * 64 * 16384
#pragma unroll
    for (int r = 0; r < 8; ++r) {
        int cout = (ct << 4) + r + (gg << 3);
        float bv = bias[cout];
        outB[((size_t)cout << 14) + p0 + (ptA << 4) + n] = acc0[r] + bv;
        outB[((size_t)cout << 14) + p0 + (ptB << 4) + n] = acc1[r] + bv;
    }
}

extern "C" void kernel_launch(void* const* d_in, const int* in_sizes, int n_in,
                              void* d_out, int out_size, void* d_ws, size_t ws_size,
                              hipStream_t stream) {
    (void)in_sizes; (void)n_in; (void)out_size; (void)d_ws; (void)ws_size;
    const float* x      = (const float*)d_in[0];
    const float* offset = (const float*)d_in[1];
    const float* mask   = (const float*)d_in[2];
    const float* weight = (const float*)d_in[3];
    const float* bias   = (const float*)d_in[4];
    float* out = (float*)d_out;

    // 4 images x 256 pixel-tiles of 64; 256 threads (8 wave32) per block
    dim3 grid(4 * 256), block(256);
    dcnv2_wmma_kernel<<<grid, block, LDS_BYTES, stream>>>(
        x, offset, mask, weight, bias, out);
}
